// GNS_67594195304974
// MI455X (gfx1250) — matrix operand order repaired and checked by reading, not compiled, not run on hardware
//
#include <hip/hip_runtime.h>
#include <hip/hip_bf16.h>
#include <cmath>

// Problem constants (from reference)
#define NB   200000
#define NL   400000
#define NG   20000
#define LATD 10
#define KIT  8

typedef __attribute__((ext_vector_type(16))) _Float16 v16h;
typedef __attribute__((ext_vector_type(8)))  float    v8f;

struct Net { const float *w1,*b1,*w2,*b2,*w3,*b3; };

// ---------------- WMMA helpers (16x16x32 f16 -> f32) ----------------

__device__ __forceinline__ v8f wmma32(v16h a, v16h b, v8f c) {
  // 8 args: (neg_a, A, neg_b, B, c_mod, C, reuse_a, reuse_b)
  return __builtin_amdgcn_wmma_f32_16x16x32_f16(false, a, false, b, (short)0, c,
                                                false, false);
}

// Pre-load bias into the C accumulator: every C element a lane holds belongs to
// column (lane&15), so all 8 get the same (zero-padded) bias value.
__device__ __forceinline__ v8f biasC(const float* bias, int lane) {
  float bc = bias[lane & 15];
  v8f c;
#pragma unroll
  for (int i = 0; i < 8; ++i) c[i] = bc;
  return c;
}

// Wave-level ordering for wave-private LDS tiles: LDS ops from one wave execute
// in order (DScnt in-order); we only need to stop compiler reordering.
__device__ __forceinline__ void wave_sync() {
  asm volatile("" ::: "memory");
  __builtin_amdgcn_wave_barrier();
  asm volatile("" ::: "memory");
}

// A matrix (16x32, MxK) from row-major [16][32] LDS tile (2x ds_load_b128).
// Lane l<16: row=l, K 0..7 + 16..23 ; lane l>=16: row=l-16, K 8..15 + 24..31.
__device__ __forceinline__ v16h load_A(const _Float16* X, int lane) {
  int row = lane & 15;
  int kb  = (lane < 16) ? 0 : 8;
  v16h a;
#pragma unroll
  for (int i = 0; i < 8; ++i) a[i]     = X[row*32 + kb + i];
#pragma unroll
  for (int i = 0; i < 8; ++i) a[8 + i] = X[row*32 + 16 + kb + i];
  return a;
}

// Store C (leaky_relu 0.01, bias already accumulated) into row-major [16][32]
// LDS tile (cols 0..15). leaky(x) = max(x, 0.01x).
__device__ __forceinline__ void store_act(_Float16* H, v8f c, int lane) {
  int col = lane & 15;
  int mb  = (lane < 16) ? 0 : 8;
#pragma unroll
  for (int r = 0; r < 8; ++r) {
    float val = fmaxf(c[r], 0.01f * c[r]);
    H[(mb + r)*32 + col] = (_Float16)val;
  }
}

// 3-layer MLP on one 16-row tile using pre-packed f16 weight fragments.
// wp: 3*512 halves (32x16 padded fragment block per layer, per-lane order)
// bp: 3*16 padded f32 biases. sH: wave-private 16x32 half scratch, cols 16..31
// pre-zeroed. Returns layer-3 accumulator WITH bias included.
__device__ __forceinline__ v8f mlp3p(v16h aNet, const _Float16* wp, const float* bp,
                                     _Float16* sH, int lane) {
  v8f c1 = wmma32(aNet, *(const v16h*)(wp + lane*16), biasC(bp, lane));
  store_act(sH, c1, lane);
  wave_sync();
  v16h a = load_A(sH, lane);
  wave_sync();
  v8f c2 = wmma32(a, *(const v16h*)(wp + 512 + lane*16), biasC(bp + 16, lane));
  store_act(sH, c2, lane);
  wave_sync();
  a = load_A(sH, lane);
  wave_sync();
  return wmma32(a, *(const v16h*)(wp + 1024 + lane*16), biasC(bp + 32, lane));
}

__device__ __forceinline__ float waveSum(float v) {
#pragma unroll
  for (int off = 16; off > 0; off >>= 1) v += __shfl_xor(v, off, 32);
  return v;
}

// ---------------- weight pre-packing ----------------
// Tile t = (n*KIT + k)*3 + l ; n: 0=phi 1=L_theta 2=L_v 3=L_m.
// Each block (32 threads) packs one layer: 512 f16 fragment entries + 16 biases.
__global__ void k_pack(Net ph, Net lt, Net lv, Net lm,
                       _Float16* wpack, float* bpack) {
  int tile = blockIdx.x;
  int lane = threadIdx.x;
  int l  = tile % 3;
  int nk = tile / 3;
  int k  = nk % KIT;
  int n  = nk / KIT;
  Net net = (n == 0) ? ph : (n == 1) ? lt : (n == 2) ? lv : lm;
  int dout3 = (n == 3) ? 10 : 1;
  const float* W; const float* B; int din, dout;
  if (l == 0)      { W = net.w1 + (size_t)k*150;      B = net.b1 + (size_t)k*10;    din = 15; dout = 10; }
  else if (l == 1) { W = net.w2 + (size_t)k*100;      B = net.b2 + (size_t)k*10;    din = 10; dout = 10; }
  else             { W = net.w3 + (size_t)k*10*dout3; B = net.b3 + (size_t)k*dout3; din = 10; dout = dout3; }
  // B fragment layout: lane l<16: col=l, K 0..7 + 16..23 ; lane l>=16: col=l-16, K 8..15 + 24..31
  int col = lane & 15;
  int kb  = (lane < 16) ? 0 : 8;
  _Float16* dst = wpack + (size_t)tile*512 + lane*16;
  for (int i = 0; i < 16; ++i) {
    int kk = (i < 8) ? kb + i : 16 + kb + (i - 8);
    float w = (kk < din && col < dout) ? W[kk*dout + col] : 0.f;
    dst[i] = (_Float16)w;
  }
  if (lane < 16) bpack[tile*16 + lane] = (lane < dout) ? B[lane] : 0.f;
}

// ---------------- init kernels ----------------

__global__ void k_zero_scal(float* scal) {
  if (threadIdx.x < 32) scal[threadIdx.x] = 0.f;
}

__global__ void k_init_bus(const float* buses, float* m, float* v, float* th,
                           float* nong, float* pgbus, float* phi,
                           float* dpacc, float* dqacc, float* qgacc, float* scal) {
  int i = blockIdx.x*blockDim.x + threadIdx.x;
  float pd = 0.f, gs = 0.f;
  if (i < NB) {
    v[i] = 1.f; th[i] = 0.f; nong[i] = 1.f; pgbus[i] = 0.f;
    phi[i] = 0.f; dpacc[i] = 0.f; dqacc[i] = 0.f; qgacc[i] = 0.f;
#pragma unroll
    for (int j = 0; j < LATD; ++j) m[(size_t)i*LATD + j] = 0.f;
    pd = buses[i*6 + 1];
    gs = buses[i*6 + 3];
  }
  pd = waveSum(pd); gs = waveSum(gs);
  if ((threadIdx.x & 31) == 0) {
    atomicAdd(&scal[0], pd);   // pd_sum
    atomicAdd(&scal[1], gs);   // gs_sum
  }
}

__global__ void k_init_gen(const float* gens, float* v, float* nong,
                           float* pgbus, float* scal) {
  int i = blockIdx.x*blockDim.x + threadIdx.x;
  float ss = 0.f, sn = 0.f, sx = 0.f;
  if (i < NG) {
    int g = (int)gens[i*6 + 0] - 1;
    v[g] = gens[i*6 + 1];
    nong[g] = 0.f;
    float pset = gens[i*6 + 2];
    pgbus[g] = pset;                 // generators[:,5] initial == pg_set
    ss = pset; sn = gens[i*6 + 3]; sx = gens[i*6 + 4];
  }
  ss = waveSum(ss); sn = waveSum(sn); sx = waveSum(sx);
  if ((threadIdx.x & 31) == 0) {
    atomicAdd(&scal[2], ss);   // s_set
    atomicAdd(&scal[3], sn);   // s_min
    atomicAdd(&scal[4], sx);   // s_max
  }
}

__global__ void k_init_pq(const float* buses, const float* v, const float* pgbus,
                          float* dp, float* dq) {
  int i = blockIdx.x*blockDim.x + threadIdx.x;
  if (i >= NB) return;
  float vi = v[i], v2 = vi*vi;
  dp[i] = -buses[i*6 + 1] - buses[i*6 + 3]*v2 + pgbus[i];
  dq[i] = -buses[i*6 + 2] - buses[i*6 + 4]*v2;   // buses[:,5] == 0 initially
}

// ---------------- per-iteration kernels ----------------

// phi MLP per line tile of 16, scatter-add into phi[fi]. 8 waves/block.
__global__ void __launch_bounds__(256) k_phi(const float* lines, const float* m,
                                             float* phi,
                                             const _Float16* wp, const float* bp) {
  __shared__ _Float16 sA[8][512];
  __shared__ _Float16 sH[8][512];
  __shared__ float    sOut[8][16];
  int wave = threadIdx.x >> 5, lane = threadIdx.x & 31;
  int base = (blockIdx.x*8 + wave) * 16;
  // Row-owner fill: lane<16 owns one row, contiguous loads, no per-col masks.
  if (lane < 16) {
    int line = base + lane;
    const float* Lrow = lines + (size_t)line*7;
    int ti = (int)Lrow[1] - 1;
    _Float16* dst = &sA[wave][lane*32];
#pragma unroll
    for (int j = 0; j < LATD; ++j) dst[j] = (_Float16)m[(size_t)ti*LATD + j];
#pragma unroll
    for (int j = 0; j < 5; ++j) dst[LATD + j] = (_Float16)Lrow[2 + j];
#pragma unroll
    for (int j = 15; j < 32; ++j) dst[j] = (_Float16)0.f;
  }
  {
    int row = lane & 15, c0 = (lane < 16) ? 0 : 16;
#pragma unroll
    for (int c = 0; c < 16; ++c) sH[wave][row*32 + c0 + c] = (_Float16)0.f;
  }
  wave_sync();
  v16h a = load_A(&sA[wave][0], lane);
  v8f c3 = mlp3p(a, wp, bp, &sH[wave][0], lane);   // bias included
  if ((lane & 15) == 0) {
    int mb = (lane < 16) ? 0 : 8;
#pragma unroll
    for (int r = 0; r < 8; ++r) sOut[wave][mb + r] = c3[r];
  }
  wave_sync();
  if (lane < 16) {
    int l2 = base + lane;
    int fi = (int)lines[(size_t)l2*7 + 0] - 1;
    atomicAdd(&phi[fi], sOut[wave][lane]);
  }
}

// Per-bus update: L_theta, L_v, L_m MLPs on one tile of 16 buses. 4 waves/block.
// Also consumes-and-zeroes phi for the next iteration.
__global__ void __launch_bounds__(128) k_bus(float* m, float* v, float* theta,
    const float* dp, const float* dq, float* phi, const float* nong,
    float* scal,
    const _Float16* wpt, const float* bpt,
    const _Float16* wpv, const float* bpv,
    const _Float16* wpm, const float* bpm) {
  __shared__ _Float16 sNet[4][512];
  __shared__ _Float16 sH[4][512];
  __shared__ float    sC[4][256];
  __shared__ float    sOut[4][16];
  int wave = threadIdx.x >> 5, lane = threadIdx.x & 31;
  int base = (blockIdx.x*4 + wave) * 16;
  if (lane < 16) {
    int bus = base + lane;
    _Float16* dst = &sNet[wave][lane*32];
    dst[0] = (_Float16)v[bus];
    dst[1] = (_Float16)theta[bus];
    dst[2] = (_Float16)dp[bus];
    dst[3] = (_Float16)dq[bus];
#pragma unroll
    for (int j = 0; j < LATD; ++j) dst[4 + j] = (_Float16)m[(size_t)bus*LATD + j];
    float ph = phi[bus];
    dst[14] = (_Float16)ph;
    phi[bus] = 0.f;                 // ready for next iteration's scatter
#pragma unroll
    for (int j = 15; j < 32; ++j) dst[j] = (_Float16)0.f;
  }
  {
    int row = lane & 15, c0 = (lane < 16) ? 0 : 16;
#pragma unroll
    for (int c = 0; c < 16; ++c) sH[wave][row*32 + c0 + c] = (_Float16)0.f;
  }
  wave_sync();
  v16h aNet = load_A(&sNet[wave][0], lane);

  // L_theta: theta += out
  v8f ct = mlp3p(aNet, wpt, bpt, &sH[wave][0], lane);
  if ((lane & 15) == 0) {
    int mb = (lane < 16) ? 0 : 8;
#pragma unroll
    for (int r = 0; r < 8; ++r) sOut[wave][mb + r] = ct[r];
  }
  wave_sync();
  if (lane < 16) theta[base + lane] += sOut[wave][lane];
  wave_sync();

  // L_v: v += out where non-gen; accumulate sum(v^2)
  v8f cv = mlp3p(aNet, wpv, bpv, &sH[wave][0], lane);
  if ((lane & 15) == 0) {
    int mb = (lane < 16) ? 0 : 8;
#pragma unroll
    for (int r = 0; r < 8; ++r) sOut[wave][mb + r] = cv[r];
  }
  wave_sync();
  float vv = 0.f;
  if (lane < 16) {
    int b2 = base + lane;
    float nvv = v[b2];
    if (nong[b2] != 0.f) nvv += sOut[wave][lane];
    v[b2] = nvv;
    vv = nvv*nvv;
  }
  vv = waveSum(vv);
  if (lane == 0) atomicAdd(&scal[6], vv);
  wave_sync();

  // L_m: m += out (16 rows x 10 cols; padded cols are exactly zero)
  v8f cm = mlp3p(aNet, wpm, bpm, &sH[wave][0], lane);
  {
    int col = lane & 15;
    int mb  = (lane < 16) ? 0 : 8;
#pragma unroll
    for (int r = 0; r < 8; ++r)
      sC[wave][(mb + r)*16 + col] = cm[r];
  }
  wave_sync();
  for (int idx = lane; idx < 160; idx += 32) {
    int r = idx / 10, j = idx % 10;
    m[(size_t)(base + r)*LATD + j] += sC[wave][r*16 + j];
  }
}

// Per-line physics: p_joule reduction + qg/dp/dq line scatters.
__global__ void k_line(const float* lines, const float* v, const float* th,
                       float* dpacc, float* dqacc, float* qgacc, float* scal) {
  int i = blockIdx.x*blockDim.x + threadIdx.x;
  float pj = 0.f;
  if (i < NL) {
    const float* L = lines + (size_t)i*7;
    int fi = (int)L[0] - 1, ti = (int)L[1] - 1;
    float r = L[2], x = L[3], bsh = L[4], tau = L[5], thsh = L[6];
    float y = 1.f / sqrtf(r*r + x*x);
    float delta = atan2f(r, x);
    float vi = v[fi], vj = v[ti], thi = th[fi], thj = th[ti];
    float sd = sinf(delta), cd = cosf(delta);
    float a1 = thi - thj - delta - thsh;
    float a2 = thj - thi - delta + thsh;
    float a3 = thj - thi - delta - thsh;
    float vit = vi / tau, vjt = vj / tau;
    float com = vi*vj*y/tau;
    pj = fabsf(com*(sinf(a1) + sinf(a2)) + vit*vit*y*sd + vj*vj*y*sd);
    float qg_line = -(-com*cosf(a1) + vit*vit*(y*cd - 0.5f*bsh))
                    -(-com*cosf(a3) + vj*vj*y*sd - 0.5f*bsh);
    float dp_line = com*sinf(a1) + vit*vit*y*sd
                  + (com*sinf(a3) + vjt*vjt*y*sd);
    float dq_line = -com*cosf(a1) + vit*vit*(y*cd - 0.5f*bsh)
                  + (-com*cosf(a3) + vj*vj*y*cd - 0.5f*bsh);
    atomicAdd(&qgacc[fi], qg_line);
    atomicAdd(&dpacc[fi], dp_line);
    atomicAdd(&dqacc[fi], dq_line);
  }
  pj = waveSum(pj);
  if ((threadIdx.x & 31) == 0) atomicAdd(&scal[5], pj);
}

__global__ void k_scalar(float* scal) {
  if (threadIdx.x == 0 && blockIdx.x == 0) {
    float pd_sum = scal[0], gs_sum = scal[1];
    float s_set = scal[2], s_min = scal[3], s_max = scal[4];
    float p_global = scal[6]*gs_sum + pd_sum + scal[5];
    float lam = (p_global < s_set)
        ? (p_global - s_min) / (2.f*(s_set - s_min))
        : (p_global - 2.f*s_set + s_max) / (2.f*(s_max - s_set));
    scal[7] = lam;
    scal[5] = 0.f;     // p_joule for next iteration
    scal[6] = 0.f;     // sum(v^2) for next iteration
  }
}

__global__ void k_gen(const float* gens, float* pgbus, const float* scal) {
  int i = blockIdx.x*blockDim.x + threadIdx.x;
  if (i >= NG) return;
  float lam = scal[7];
  float pset = gens[i*6 + 2], pmin = gens[i*6 + 3], pmax = gens[i*6 + 4];
  float pg = (lam < 0.5f) ? (pmin + 2.f*(pset - pmin)*lam)
                          : (2.f*pset - pmax + 2.f*(pmax - pset)*lam);
  int g = (int)gens[i*6 + 0] - 1;
  pgbus[g] = pg;
}

// Consumes-and-zeroes the three scatter accumulators.
__global__ void k_finish(const float* buses, const float* v, const float* pgbus,
                         float* dpacc, float* dqacc, float* qgacc,
                         float* dp, float* dq, float* scal, float wgt) {
  int i = blockIdx.x*blockDim.x + threadIdx.x;
  float loc = 0.f;
  if (i < NB) {
    float vi = v[i], v2 = vi*vi;
    float Pd = buses[i*6 + 1], Qd = buses[i*6 + 2];
    float Gs = buses[i*6 + 3], Bs = buses[i*6 + 4];
    float qa = qgacc[i], da = dpacc[i], db = dqacc[i];
    qgacc[i] = 0.f; dpacc[i] = 0.f; dqacc[i] = 0.f;
    float qg_new = Qd - Bs*v2 + qa;
    float dpn = -Pd - Gs*v2 + pgbus[i] + da;
    float dqn = qg_new - Qd - Bs*v2 + db;
    dp[i] = dpn; dq[i] = dqn;
    loc = dpn*dpn + dqn*dqn;
  }
  loc = waveSum(loc);
  if ((threadIdx.x & 31) == 0)
    atomicAdd(&scal[8], wgt * loc / (float)NB);
}

__global__ void k_output(const float* v, const float* th, const float* scal,
                         float* out) {
  int i = blockIdx.x*blockDim.x + threadIdx.x;
  if (i < NB) { out[i] = v[i]; out[NB + i] = th[i]; }
  if (i == 0) out[2*NB] = scal[8];
}

// ---------------- host launch ----------------

extern "C" void kernel_launch(void* const* d_in, const int* in_sizes, int n_in,
                              void* d_out, int out_size, void* d_ws, size_t ws_size,
                              hipStream_t stream) {
  (void)in_sizes; (void)n_in; (void)out_size; (void)ws_size;
  const float* buses = (const float*)d_in[0];
  const float* lines = (const float*)d_in[1];
  const float* gens  = (const float*)d_in[2];
  // params pytree: dict keys sorted -> L_m, L_theta, L_v, phi; each (w1,b1,w2,b2,w3,b3)
  Net Lm = { (const float*)d_in[3],  (const float*)d_in[4],  (const float*)d_in[5],
             (const float*)d_in[6],  (const float*)d_in[7],  (const float*)d_in[8] };
  Net Lt = { (const float*)d_in[9],  (const float*)d_in[10], (const float*)d_in[11],
             (const float*)d_in[12], (const float*)d_in[13], (const float*)d_in[14] };
  Net Lv = { (const float*)d_in[15], (const float*)d_in[16], (const float*)d_in[17],
             (const float*)d_in[18], (const float*)d_in[19], (const float*)d_in[20] };
  Net Ph = { (const float*)d_in[21], (const float*)d_in[22], (const float*)d_in[23],
             (const float*)d_in[24], (const float*)d_in[25], (const float*)d_in[26] };
  float* out = (float*)d_out;

  float* ws    = (float*)d_ws;
  float* wm    = ws;                                  // NB*LATD
  float* wphi  = wm + (size_t)NB*LATD;                // NB
  float* wv    = wphi + NB;                           // NB
  float* wth   = wv + NB;                             // NB
  float* wdp   = wth + NB;                            // NB
  float* wdq   = wdp + NB;                            // NB
  float* wdpa  = wdq + NB;                            // NB
  float* wdqa  = wdpa + NB;                           // NB
  float* wqga  = wdqa + NB;                           // NB
  float* wpg   = wqga + NB;                           // NB (Pg scattered at gen buses)
  float* wng   = wpg + NB;                            // NB (non-gen mask)
  float* wscal = wng + NB;                            // 32 scalars
  // packed weights: 96 layer tiles * 512 halves (32B-aligned), then 96*16 biases
  _Float16* wpackh = (_Float16*)(wscal + 32);
  float*    wbp    = (float*)(wpackh + (size_t)96*512);

  const int BG = (NB + 255) / 256;   // 782
  const int GG = (NG + 255) / 256;   // 79
  const int LG = (NL + 255) / 256;   // 1563

  k_zero_scal<<<1, 32, 0, stream>>>(wscal);
  k_pack     <<<96, 32, 0, stream>>>(Ph, Lt, Lv, Lm, wpackh, wbp);
  k_init_bus <<<BG, 256, 0, stream>>>(buses, wm, wv, wth, wng, wpg,
                                      wphi, wdpa, wdqa, wqga, wscal);
  k_init_gen <<<GG, 256, 0, stream>>>(gens, wv, wng, wpg, wscal);
  k_init_pq  <<<BG, 256, 0, stream>>>(buses, wv, wpg, wdp, wdq);

  for (int k = 0; k < KIT; ++k) {
    // per-(net,k) packed blocks: tile base = (n*KIT + k)*3
    const _Float16* wp_ph = wpackh + (size_t)((0*KIT + k)*3)*512;
    const float*    bp_ph = wbp    + (size_t)((0*KIT + k)*3)*16;
    const _Float16* wp_t  = wpackh + (size_t)((1*KIT + k)*3)*512;
    const float*    bp_t  = wbp    + (size_t)((1*KIT + k)*3)*16;
    const _Float16* wp_v  = wpackh + (size_t)((2*KIT + k)*3)*512;
    const float*    bp_v  = wbp    + (size_t)((2*KIT + k)*3)*16;
    const _Float16* wp_m  = wpackh + (size_t)((3*KIT + k)*3)*512;
    const float*    bp_m  = wbp    + (size_t)((3*KIT + k)*3)*16;

    k_phi<<<NL/16/8, 256, 0, stream>>>(lines, wm, wphi, wp_ph, bp_ph);   // 3125
    k_bus<<<NB/16/4, 128, 0, stream>>>(wm, wv, wth, wdp, wdq, wphi,      // 3125
                                       wng, wscal, wp_t, bp_t, wp_v, bp_v,
                                       wp_m, bp_m);
    k_line<<<LG, 256, 0, stream>>>(lines, wv, wth, wdpa, wdqa, wqga, wscal);
    k_scalar<<<1, 1, 0, stream>>>(wscal);
    k_gen<<<GG, 256, 0, stream>>>(gens, wpg, wscal);
    float wgt = powf(0.9f, (float)(KIT - k));
    k_finish<<<BG, 256, 0, stream>>>(buses, wv, wpg, wdpa, wdqa, wqga,
                                     wdp, wdq, wscal, wgt);
  }
  k_output<<<BG, 256, 0, stream>>>(wv, wth, wscal, out);
}
